// SelfAttention_9655086481988
// MI455X (gfx1250) — compile-verified, matrix operand order checked by reading, not run on hardware
//
#include <hip/hip_runtime.h>
#include <hip/hip_bf16.h>

typedef __bf16 bf16_t;
typedef __attribute__((ext_vector_type(16))) __bf16 v16bf;
typedef __attribute__((ext_vector_type(8)))  __bf16 v8bf;
typedef __attribute__((ext_vector_type(8)))  float  v8f;
typedef int v4i_vs __attribute__((vector_size(16)));   // matches builtin proto

#define N_EMBD 768
#define HEAD   64
#define TSEQ   2048
#define NBATCH 8

#ifndef __has_builtin
#define __has_builtin(x) 0
#endif
#if __has_builtin(__builtin_amdgcn_global_load_async_to_lds_b128)
#define HAVE_ASYNC_LDS 1
#else
#define HAVE_ASYNC_LDS 0
#endif

#define AS1 __attribute__((address_space(1)))
#define AS3 __attribute__((address_space(3)))

// ---- helpers -------------------------------------------------------------

__device__ __forceinline__ v16bf cat16(v8bf lo, v8bf hi) {
  return __builtin_shufflevector(lo, hi, 0,1,2,3,4,5,6,7,8,9,10,11,12,13,14,15);
}

// Fragment stripe for this lane: halves [p..p+7] and [p+16..p+23]
__device__ __forceinline__ v16bf load_frag(const bf16_t* p) {
  v8bf lo = *(const v8bf*)p;
  v8bf hi = *(const v8bf*)(p + 16);
  return cat16(lo, hi);
}

__device__ __forceinline__ v8f wmma_bf16(v16bf a, v16bf b, v8f c) {
  return __builtin_amdgcn_wmma_f32_16x16x32_bf16(
      false, a, false, b, (short)0, c, false, false);
}

#if HAVE_ASYNC_LDS
__device__ __forceinline__ void async_b128(const bf16_t* g, bf16_t* l) {
  __builtin_amdgcn_global_load_async_to_lds_b128(
      (AS1 v4i_vs*)g, (AS3 v4i_vs*)l, 0, 0);
}
template <int N>
__device__ __forceinline__ void wait_async() {
#if __has_builtin(__builtin_amdgcn_s_wait_asynccnt)
  __builtin_amdgcn_s_wait_asynccnt(N);
#else
  asm volatile("s_wait_asynccnt %0" :: "i"(N) : "memory");
#endif
}
#endif

// ---- kernel 1: W transpose + f32->bf16 (scale folded into Wq) ------------

__global__ void wt_prep(const float* __restrict__ Wq, const float* __restrict__ Wk,
                        const float* __restrict__ Wv, bf16_t* __restrict__ wt) {
  int idx = blockIdx.x * blockDim.x + threadIdx.x;   // 3*64*768 total
  if (idx >= 3 * HEAD * N_EMBD) return;
  int m   = idx / (HEAD * N_EMBD);
  int rem = idx - m * (HEAD * N_EMBD);
  int h   = rem / N_EMBD;
  int c   = rem - h * N_EMBD;
  const float* W = (m == 0) ? Wq : (m == 1) ? Wk : Wv;
  float v = W[c * HEAD + h];
  if (m == 0) v *= 0.125f;                 // 1/sqrt(HEAD), exact in bf16
  wt[idx] = (bf16_t)v;                     // wt[m][h][c]
}

// ---- kernel 2: QKV projection, 16 rows/block, waves 0/1/2 -> q/k/v -------
// q,k stored row-major [B*T][64]; v stored TRANSPOSED as vT[B][64][2048].

#define XS_STRIDE 776                      // 768 + 8 halves: conflict-free rows

__global__ __launch_bounds__(96)
void qkv_proj(const float* __restrict__ x, const bf16_t* __restrict__ wt,
              bf16_t* __restrict__ qkv) {
  __shared__ bf16_t xs[16 * XS_STRIDE];
  const int tid  = threadIdx.x;
  const int lane = tid & 31;
  const int wave = tid >> 5;               // 0:q 1:k 2:v
  const long rowBase = (long)blockIdx.x * 16;   // row in [0, B*T)

  for (int i = tid; i < 16 * 768 / 4; i += 96) {
    int r  = i / 192;
    int c4 = i - r * 192;
    const float4 f = ((const float4*)(x + (rowBase + r) * N_EMBD))[c4];
    bf16_t* dst = xs + r * XS_STRIDE + c4 * 4;
    dst[0] = (bf16_t)f.x; dst[1] = (bf16_t)f.y;
    dst[2] = (bf16_t)f.z; dst[3] = (bf16_t)f.w;
  }
  __syncthreads();

  const int n    = lane & 15;
  const int koff = (lane >> 4) * 8;
  const int mhi  = (lane >> 4) * 8;
  const bf16_t* wmat = wt + (long)wave * (HEAD * N_EMBD);

  v8f acc[4] = {};
  for (int cb = 0; cb < N_EMBD; cb += 32) {
    v16bf afrag = load_frag(xs + n * XS_STRIDE + cb + koff);
#pragma unroll
    for (int t = 0; t < 4; ++t) {
      v16bf bfrag = load_frag(wmat + (t * 16 + n) * N_EMBD + cb + koff);
      acc[t] = wmma_bf16(afrag, bfrag, acc[t]);
    }
  }

  if (wave < 2) {
    // q / k row-major
    bf16_t* out = qkv + (long)wave * (NBATCH * TSEQ * HEAD) + rowBase * HEAD;
#pragma unroll
    for (int t = 0; t < 4; ++t)
#pragma unroll
      for (int r = 0; r < 8; ++r)
        out[(long)(r + mhi) * HEAD + t * 16 + n] = (bf16_t)acc[t][r];
  } else {
    // v transposed: vT[b][h][t]
    const long bb = rowBase >> 11;
    const int  t0 = (int)(rowBase & 2047);
    bf16_t* outv = qkv + 2L * NBATCH * TSEQ * HEAD + bb * (long)(HEAD * TSEQ);
#pragma unroll
    for (int t = 0; t < 4; ++t)
#pragma unroll
      for (int r = 0; r < 8; ++r)
        outv[(long)(t * 16 + n) * TSEQ + t0 + r + mhi] = (bf16_t)acc[t][r];
  }
}

// ---- kernel 3: causal flash attention, 64 q-rows/block, 4 waves,
//                double-buffered async K/V staging ------------------------

#define KT_STRIDE 72   // 64+8 halves
#define VT_STRIDE 40   // 32+8 halves
#define PS_STRIDE 40

__global__ __launch_bounds__(128)
void attn(const bf16_t* __restrict__ qkv, float* __restrict__ out) {
  __shared__ bf16_t kt[2][32 * KT_STRIDE];       // K block, row-major [s][h]
  __shared__ bf16_t vt[2][64 * VT_STRIDE];       // V block transposed [h][s]
  __shared__ bf16_t ps[4][16 * PS_STRIDE];       // per-wave P scratch [m][s]

  const int tid  = threadIdx.x;
  const int lane = tid & 31;
  const int wave = tid >> 5;
  const int b    = blockIdx.x >> 5;
  const int rb   = blockIdx.x & 31;
  const int qblk = rb * 64;
  const int q0   = qblk + wave * 16;

  const bf16_t* qm  = qkv;
  const bf16_t* km  = qkv + (long)NBATCH * TSEQ * HEAD;
  const bf16_t* vmt = qkv + 2L * NBATCH * TSEQ * HEAD + (long)b * (HEAD * TSEQ);
  const long base   = (long)b * TSEQ;

  const int n    = lane & 15;
  const int hi   = lane >> 4;
  const int koff = hi * 8;

  v16bf qfrag[2];
  {
    const bf16_t* qrow = qm + (base + q0 + n) * HEAD;
    qfrag[0] = load_frag(qrow + koff);
    qfrag[1] = load_frag(qrow + 32 + koff);
  }

  v8f accO[4] = {};
  float mrow[8], lrow[8];
#pragma unroll
  for (int r = 0; r < 8; ++r) { mrow[r] = -3.0e38f; lrow[r] = 0.0f; }

  const int nIter = (qblk + 64) / 32;

  // stage one tile (K rows [s0..s0+31], vT cols [s0..s0+31]); 4 asyncs/wave
  auto stage = [&](int it, int buf) {
    const bf16_t* kg = km + (base + (long)it * 32) * HEAD;
    const bf16_t* vg = vmt + it * 32;
    bf16_t* ktb = kt[buf];
    bf16_t* vtb = vt[buf];
#if HAVE_ASYNC_LDS
    for (int i = tid; i < 256; i += 128) {
      int r = i >> 3, c8 = (i & 7) * 8;
      async_b128(kg + r * HEAD + c8, ktb + r * KT_STRIDE + c8);
    }
    for (int i = tid; i < 256; i += 128) {
      int h = i >> 2, c8 = (i & 3) * 8;
      async_b128(vg + (long)h * TSEQ + c8, vtb + h * VT_STRIDE + c8);
    }
#else
    for (int i = tid; i < 256; i += 128) {
      int r = i >> 3, c8 = (i & 7) * 8;
      *(v8bf*)(ktb + r * KT_STRIDE + c8) = *(const v8bf*)(kg + r * HEAD + c8);
    }
    for (int i = tid; i < 256; i += 128) {
      int h = i >> 2, c8 = (i & 3) * 8;
      *(v8bf*)(vtb + h * VT_STRIDE + c8) = *(const v8bf*)(vg + (long)h * TSEQ + c8);
    }
#endif
  };

  stage(0, 0);                             // prologue: tile 0 in flight

  for (int it = 0; it < nIter; ++it) {
    const int s0  = it * 32;
    const int buf = it & 1;

#if HAVE_ASYNC_LDS
    if (it + 1 < nIter) {
      stage(it + 1, buf ^ 1);              // prefetch next tile
      wait_async<4>();                     // tile `it` resident (in-order)
    } else {
      wait_async<0>();
    }
#else
    if (it + 1 < nIter) stage(it + 1, buf ^ 1);
#endif
    __syncthreads();                       // tile `it` visible to all waves

    if (s0 <= q0 + 15) {
      const bf16_t* ktb = kt[buf];
      const bf16_t* vtb = vt[buf];

      v8f S[2] = {};
#pragma unroll
      for (int st = 0; st < 2; ++st)
#pragma unroll
        for (int kk = 0; kk < 2; ++kk) {
          v16bf bfrag = load_frag(ktb + (st * 16 + n) * KT_STRIDE + kk * 32 + koff);
          S[st] = wmma_bf16(qfrag[kk], bfrag, S[st]);
        }

      if (s0 + 31 > q0) {                  // diagonal block: causal mask
#pragma unroll
        for (int st = 0; st < 2; ++st)
#pragma unroll
          for (int r = 0; r < 8; ++r) {
            int qr = q0 + r + hi * 8;
            int sc = s0 + st * 16 + n;
            if (sc > qr) S[st][r] = -3.0e38f;
          }
      }

      float mnew[8], alpha[8];
#pragma unroll
      for (int r = 0; r < 8; ++r) {
        float mx = fmaxf(S[0][r], S[1][r]);
#pragma unroll
        for (int sh = 8; sh >= 1; sh >>= 1) mx = fmaxf(mx, __shfl_xor(mx, sh, 32));
        mnew[r]  = fmaxf(mrow[r], mx);
        alpha[r] = __expf(mrow[r] - mnew[r]);
      }

      bf16_t* pw = &ps[wave][0];
#pragma unroll
      for (int r = 0; r < 8; ++r) {
        float p0 = __expf(S[0][r] - mnew[r]);
        float p1 = __expf(S[1][r] - mnew[r]);
        float rs = p0 + p1;
#pragma unroll
        for (int sh = 8; sh >= 1; sh >>= 1) rs += __shfl_xor(rs, sh, 32);
        lrow[r] = lrow[r] * alpha[r] + rs;
        mrow[r] = mnew[r];
        int M = r + hi * 8;
        pw[M * PS_STRIDE + n]      = (bf16_t)p0;
        pw[M * PS_STRIDE + n + 16] = (bf16_t)p1;
      }

#pragma unroll
      for (int t = 0; t < 4; ++t)
#pragma unroll
        for (int r = 0; r < 8; ++r) accO[t][r] *= alpha[r];

      // re-read P as A-fragment (same-wave LDS is in-order; block reordering)
      asm volatile("" ::: "memory");
      v16bf pfrag = load_frag(pw + n * PS_STRIDE + koff);
#pragma unroll
      for (int t = 0; t < 4; ++t) {
        v16bf bfrag = load_frag(vtb + (t * 16 + n) * VT_STRIDE + koff);
        accO[t] = wmma_bf16(pfrag, bfrag, accO[t]);
      }
    }

    __syncthreads();                       // done reading buf before overwrite
  }

  float* orow = out + (base + q0) * HEAD;
#pragma unroll
  for (int t = 0; t < 4; ++t)
#pragma unroll
    for (int r = 0; r < 8; ++r) {
      int M = r + hi * 8;
      orow[(long)M * HEAD + t * 16 + n] = accO[t][r] / lrow[r];
    }
}

// ---- launcher ------------------------------------------------------------

extern "C" void kernel_launch(void* const* d_in, const int* in_sizes, int n_in,
                              void* d_out, int out_size, void* d_ws, size_t ws_size,
                              hipStream_t stream) {
  const float* x  = (const float*)d_in[0];
  const float* Wk = (const float*)d_in[1];
  const float* Wq = (const float*)d_in[2];
  const float* Wv = (const float*)d_in[3];

  bf16_t* wt  = (bf16_t*)d_ws;                         // [3][64][768]  (q,k,v)
  bf16_t* qkv = wt + 3 * HEAD * N_EMBD;                // q,k: [B*T][64]; vT: [B][64][2048]

  wt_prep<<<(3 * HEAD * N_EMBD + 255) / 256, 256, 0, stream>>>(Wq, Wk, Wv, wt);
  qkv_proj<<<NBATCH * TSEQ / 16, 96, 0, stream>>>(x, wt, qkv);
  attn<<<NBATCH * (TSEQ / 64), 128, 0, stream>>>(qkv, (float*)d_out);
}
